// WindowAttention3D02_90065464197203
// MI455X (gfx1250) — matrix operand order repaired and checked by reading, not compiled, not run on hardware
//
#include <hip/hip_runtime.h>
#include <hip/hip_bf16.h>

// ---- problem constants (match reference) ----
#define B_    256
#define N1_   384
#define N2_   128
#define DIM_  128
#define NH_   4
#define HD_   32
#define NW_   64
#define SCALE_F 0.17677669529663687f   // 32^-0.5

typedef __attribute__((ext_vector_type(16))) _Float16 v16h;
typedef __attribute__((ext_vector_type(8)))  _Float16 v8h;
typedef __attribute__((ext_vector_type(8)))  float    v8f;
typedef __attribute__((ext_vector_type(4)))  float    v4f;

union HV { v16h v; v8h h[2]; };

// ---------------------------------------------------------------------------
// Prep: transpose+convert weights to f16 [K][N], gather rel-pos bias (H,N1,N2)
// ---------------------------------------------------------------------------
__global__ void __launch_bounds__(256)
wa3d_prep_kernel(const float* __restrict__ qw,
                 const float* __restrict__ kvw,
                 const float* __restrict__ pw,
                 const float* __restrict__ rpb,
                 const int*   __restrict__ relidx,
                 _Float16* __restrict__ qwt,
                 _Float16* __restrict__ kvwt,
                 _Float16* __restrict__ pwt,
                 float* __restrict__ biasws) {
    int i = blockIdx.x * blockDim.x + threadIdx.x;
    if (i < DIM_ * DIM_) {                     // q_wt / proj_wt: [k][n] = W[n][k]
        int k = i >> 7, n = i & 127;
        qwt[i] = (_Float16)qw[n * DIM_ + k];
        pwt[i] = (_Float16)pw[n * DIM_ + k];
    }
    if (i < DIM_ * 2 * DIM_) {                 // kv_wt: [k][n], n<256
        int k = i >> 8, n = i & 255;
        kvwt[i] = (_Float16)kvw[n * DIM_ + k];
    }
    if (i < NH_ * N1_ * N2_) {                 // bias[h][q][k] = rpb[rel[q][k]][h]
        int h = i / (N1_ * N2_);
        int rem = i % (N1_ * N2_);
        biasws[i] = rpb[relidx[rem] * NH_ + h];
    }
}

// ---------------------------------------------------------------------------
// WMMA GEMM with A-reuse: each wave computes one 16-row x 128-col stripe
// (8 accumulator tiles).  C = A(f32|f16)[MxK] * Bt(f16)[KxN] (*scale +bias
// +resid), outputs f32 and/or f16.  ngroups = N/128 column groups.
// __launch_bounds__(256,2): allow ~512 VGPRs/wave so the 8-tile stripe never
// spills; real occupancy still ~4 waves/SIMD.
// ---------------------------------------------------------------------------
template<bool AF16>
__global__ void __launch_bounds__(256, 2)
wa3d_gemm_wmma(const void* __restrict__ Ap,
               const _Float16* __restrict__ Bt,
               const float* __restrict__ bias,
               const float* __restrict__ resid,
               float* __restrict__ outF,
               _Float16* __restrict__ outH,
               float scale, int M, int N, int K) {
    const int lane = threadIdx.x & 31;
    const int wave = threadIdx.x >> 5;
    const int ngroups = N >> 7;                 // 128-wide column groups
    const int wid = blockIdx.x * 8 + wave;
    const int tm = wid / ngroups;
    const int ng = wid % ngroups;
    if (tm * 16 >= M) return;                   // wave-uniform guard
    const int n0   = ng * 128;
    const int klo  = (lane < 16) ? 0 : 8;
    const int arow = tm * 16 + (lane & 15);

    v8f acc[8];
    {
        v8f z = {};
#pragma unroll
        for (int t = 0; t < 8; ++t) acc[t] = z;
    }

    for (int kb = 0; kb < K; kb += 32) {
        HV a;
        if constexpr (AF16) {
            const _Float16* ar = (const _Float16*)Ap + (size_t)arow * K + kb;
            a.h[0] = *(const v8h*)(ar + klo);
            a.h[1] = *(const v8h*)(ar + klo + 16);
        } else {
            const float* ar = (const float*)Ap + (size_t)arow * K + kb;
            v4f p0 = *(const v4f*)(ar + klo);
            v4f p1 = *(const v4f*)(ar + klo + 4);
            v4f p2 = *(const v4f*)(ar + klo + 16);
            v4f p3 = *(const v4f*)(ar + klo + 20);
#pragma unroll
            for (int j = 0; j < 4; ++j) {
                a.v[j]      = (_Float16)p0[j];
                a.v[4 + j]  = (_Float16)p1[j];
                a.v[8 + j]  = (_Float16)p2[j];
                a.v[12 + j] = (_Float16)p3[j];
            }
        }
        const _Float16* brow = Bt + (size_t)kb * N + n0 + (lane & 15);
        // two groups of 4 N-tiles with a scheduling fence to cap live B regs
#pragma unroll
        for (int g = 0; g < 2; ++g) {
#pragma unroll
            for (int t = 0; t < 4; ++t) {
                int nt = g * 4 + t;
                HV b;
#pragma unroll
                for (int j = 0; j < 8; ++j) {
                    b.v[j]     = brow[(size_t)(klo + j) * N + nt * 16];
                    b.v[8 + j] = brow[(size_t)(klo + 16 + j) * N + nt * 16];
                }
                acc[nt] = __builtin_amdgcn_wmma_f32_16x16x32_f16(
                              false, a.v, false, b.v, (short)0, acc[nt], false, false);
            }
            asm volatile("" ::: "memory");     // scheduling fence
        }
    }
    const int mhalf = (lane < 16) ? 0 : 8;
#pragma unroll
    for (int nt = 0; nt < 8; ++nt) {
#pragma unroll
        for (int r = 0; r < 8; ++r) {
            int row = tm * 16 + r + mhalf;
            int col = n0 + nt * 16 + (lane & 15);
            float val = acc[nt][r] * scale + bias[col];
            if (resid) val += resid[(size_t)row * N + col];
            if (outF)  outF[(size_t)row * N + col] = val;
            if (outH)  outH[(size_t)row * N + col] = (_Float16)val;
        }
    }
}

// ---------------------------------------------------------------------------
// Fused attention per (batch, head): S = Q Kt (WMMA), +bias+mask, softmax,
// write attn f32, P(f16) staged in per-wave LDS stripe, O = P V (WMMA) -> xh.
// ---------------------------------------------------------------------------
__global__ void __launch_bounds__(256, 2)
wa3d_attn_wmma(const _Float16* __restrict__ qp,
               const _Float16* __restrict__ kvp,
               const float* __restrict__ mask,
               const float* __restrict__ biasws,
               float* __restrict__ attn_out,
               _Float16* __restrict__ xh) {
    __shared__ _Float16 sKt[HD_ * N2_];        // K transposed: [feat][key]   8 KB
    __shared__ _Float16 sV [N2_ * HD_];        // V row-major:  [key][feat]   8 KB
    __shared__ _Float16 sP [8 * 16 * N2_];     // per-wave P stripes         32 KB

    const int bh = blockIdx.x;
    const int b  = bh >> 2;
    const int h  = bh & 3;
    const int widx = b & (NW_ - 1);            // b % nW (B = 4 * nW)
    const int tid = threadIdx.x;

    // --- V: async global -> LDS copy (16B chunks), tracked by ASYNCcnt ---
    for (int i = tid; i < N2_ * 4; i += 256) { // 128 rows x 4 chunks of 8 halves
        int key = i >> 2, c4 = i & 3;
        const _Float16* src = kvp + (size_t)(b * N2_ + key) * (2 * DIM_)
                            + DIM_ + h * HD_ + c4 * 8;
        unsigned long long ga = (unsigned long long)(const void*)src;
        unsigned lds = (unsigned)(size_t)&sV[key * HD_ + c4 * 8];
        asm volatile("global_load_async_to_lds_b128 %0, %1, off"
                     :: "v"(lds), "v"(ga) : "memory");
    }
    // --- Kt: on-the-fly transpose (scalar f16 stores) ---
    for (int i = tid; i < N2_ * HD_; i += 256) {
        int key = i >> 5, f = i & 31;
        sKt[f * N2_ + key] = kvp[(size_t)(b * N2_ + key) * (2 * DIM_) + h * HD_ + f];
    }
    asm volatile("s_wait_asynccnt 0" ::: "memory");
    __syncthreads();

    const int lane  = tid & 31;
    const int wave  = tid >> 5;
    const int klo   = (lane < 16) ? 0 : 8;
    const int lrow  = lane & 15;
    const int mhalf = (lane < 16) ? 0 : 8;
    _Float16* Pw = sP + wave * (16 * N2_);
    // keep bias/mask bases in SGPRs (kernargs); use 32-bit lane offsets only
    const int hoff = h * (N1_ * N2_) + lrow;       // into biasws
    const int woff = widx * (N1_ * N2_) + lrow;    // into mask

    for (int qb = wave; qb < N1_ / 16; qb += 8) {   // 3 uniform iterations/wave
        const int m0 = qb * 16;

        HV a;                                        // A-operand: Q tile row
        {
            const _Float16* qr = qp + ((size_t)(b * N1_ + m0 + lrow)) * DIM_ + h * HD_;
            a.h[0] = *(const v8h*)(qr + klo);
            a.h[1] = *(const v8h*)(qr + klo + 16);
        }

        v8f c[8];                                    // 16x128 score stripe
#pragma unroll
        for (int g = 0; g < 2; ++g) {
#pragma unroll
            for (int t = 0; t < 4; ++t) {
                int nb = g * 4 + t;
                HV bk;                               // lane = key column of Kt
                const _Float16* kr = sKt + nb * 16 + lrow;
#pragma unroll
                for (int j = 0; j < 8; ++j) {
                    bk.v[j]     = kr[(klo + j) * N2_];
                    bk.v[8 + j] = kr[(klo + 16 + j) * N2_];
                }
                v8f z = {};
                c[nb] = __builtin_amdgcn_wmma_f32_16x16x32_f16(
                            false, a.v, false, bk.v, (short)0, z, false, false);
            }
            asm volatile("" ::: "memory");           // cap live B-operand regs
        }

        // + relative-position bias + shifted-window mask
#pragma unroll
        for (int nb = 0; nb < 8; ++nb) {
#pragma unroll
            for (int r = 0; r < 8; ++r) {
                int idx = (m0 + r + mhalf) * N2_ + nb * 16;
                c[nb][r] += biasws[hoff + idx] + mask[woff + idx];
            }
        }

        // softmax over N2 per row (row lives in one 16-lane half per C layout)
        float inv[8];
#pragma unroll
        for (int r = 0; r < 8; ++r) {
            float mx = c[0][r];
#pragma unroll
            for (int nb = 1; nb < 8; ++nb) mx = fmaxf(mx, c[nb][r]);
#pragma unroll
            for (int s = 1; s < 16; s <<= 1) mx = fmaxf(mx, __shfl_xor(mx, s, 32));
            float sum = 0.f;
#pragma unroll
            for (int nb = 0; nb < 8; ++nb) {
                float e = __expf(c[nb][r] - mx);
                c[nb][r] = e;
                sum += e;
            }
#pragma unroll
            for (int s = 1; s < 16; s <<= 1) sum += __shfl_xor(sum, s, 32);
            inv[r] = 1.0f / sum;
        }

        // write attn (f32) and stage P (f16) in this wave's LDS stripe
        const int aoff = (bh * N1_ + m0) * N2_ + lrow;   // into attn_out (SGPR base)
#pragma unroll
        for (int nb = 0; nb < 8; ++nb) {
#pragma unroll
            for (int r = 0; r < 8; ++r) {
                int ml = r + mhalf;
                float p = c[nb][r] * inv[r];
                attn_out[(size_t)aoff + ml * N2_ + nb * 16] = p;
                Pw[ml * N2_ + nb * 16 + lrow] = (_Float16)p;
            }
        }
        asm volatile("s_wait_dscnt 0" ::: "memory");  // per-wave DS fence

        // O = P @ V  (K = 128 keys, two 16-wide feature tiles)
#pragma unroll
        for (int nf = 0; nf < 2; ++nf) {
            v8f o = {};
#pragma unroll
            for (int kb = 0; kb < 4; ++kb) {
                HV pa;
                const _Float16* pr = Pw + lrow * N2_ + kb * 32;
                pa.h[0] = *(const v8h*)(pr + klo);
                pa.h[1] = *(const v8h*)(pr + klo + 16);
                HV vb;                               // lane = feat column of V
                const _Float16* vr = sV + (kb * 32) * HD_ + nf * 16 + lrow;
#pragma unroll
                for (int j = 0; j < 8; ++j) {
                    vb.v[j]     = vr[(klo + j) * HD_];
                    vb.v[8 + j] = vr[(klo + 16 + j) * HD_];
                }
                o = __builtin_amdgcn_wmma_f32_16x16x32_f16(
                        false, pa.v, false, vb.v, (short)0, o, false, false);
            }
#pragma unroll
            for (int r = 0; r < 8; ++r) {
                int ml   = r + mhalf;
                int feat = h * HD_ + nf * 16 + lrow;
                xh[((size_t)(b * N1_ + m0 + ml)) * DIM_ + feat] = (_Float16)o[r];
            }
        }
    }
}

// ---------------------------------------------------------------------------
extern "C" void kernel_launch(void* const* d_in, const int* in_sizes, int n_in,
                              void* d_out, int out_size, void* d_ws, size_t ws_size,
                              hipStream_t stream) {
    const float* q      = (const float*)d_in[0];
    const float* kv     = (const float*)d_in[1];
    const float* mask   = (const float*)d_in[2];
    const float* q_w    = (const float*)d_in[3];
    const float* q_b    = (const float*)d_in[4];
    const float* kv_w   = (const float*)d_in[5];
    const float* kv_b   = (const float*)d_in[6];
    const float* proj_w = (const float*)d_in[7];
    const float* proj_b = (const float*)d_in[8];
    const float* rpb    = (const float*)d_in[9];
    const int*   relidx = (const int*)d_in[10];

    float* out_x    = (float*)d_out;                       // (B,N1,DIM)
    float* out_attn = out_x + (size_t)B_ * N1_ * DIM_;     // (B,H,N1,N2)

    // workspace layout (bytes)
    char* ws = (char*)d_ws;
    _Float16* qp   = (_Float16*)(ws);                        // 98304*128 f16 = 25165824
    _Float16* kvp  = (_Float16*)(ws + 25165824);             // 32768*256 f16 = 16777216
    _Float16* xh   = (_Float16*)(ws + 41943040);             // 98304*128 f16 = 25165824
    _Float16* qwt  = (_Float16*)(ws + 67108864);             // 128*128 f16
    _Float16* kvwt = (_Float16*)(ws + 67141632);             // 128*256 f16
    _Float16* pwt  = (_Float16*)(ws + 67207168);             // 128*128 f16
    float*    biasws = (float*)(ws + 67239936);              // 4*384*128 f32

    // 1) weights -> f16 transposed, bias gather
    wa3d_prep_kernel<<<768, 256, 0, stream>>>(q_w, kv_w, proj_w, rpb, relidx,
                                              qwt, kvwt, pwt, biasws);
    // 2) q projection (scaled) -> qp f16   (6144 row-stripes / 8 waves)
    wa3d_gemm_wmma<false><<<768, 256, 0, stream>>>(q, qwt, q_b, nullptr,
                                                   nullptr, qp, SCALE_F,
                                                   B_ * N1_, DIM_, DIM_);
    // 3) kv projection -> kvp f16 (rows = [k|v]) (2048 row-stripes x 2 groups)
    wa3d_gemm_wmma<false><<<512, 256, 0, stream>>>(kv, kvwt, kv_b, nullptr,
                                                   nullptr, kvp, 1.0f,
                                                   B_ * N2_, 2 * DIM_, DIM_);
    // 4) fused attention: writes attn f32 + xh f16
    wa3d_attn_wmma<<<B_ * NH_, 256, 0, stream>>>(qp, kvp, mask, biasws,
                                                 out_attn, xh);
    // 5) output projection + bias + residual -> out_x f32
    wa3d_gemm_wmma<true><<<768, 256, 0, stream>>>(xh, pwt, proj_b, q,
                                                  out_x, nullptr, 1.0f,
                                                  B_ * N1_, DIM_, DIM_);
}
